// HMM_76982993813765
// MI455X (gfx1250) — matrix operand order repaired and checked by reading, not compiled, not run on hardware
//
#include <hip/hip_runtime.h>
#include <hip/hip_bf16.h>
#include <math.h>

// ---------------------------------------------------------------------------
// HMM forward (log domain) on gfx1250.
//   N = 512 states, B = 64 batch, T_MAX = 256, M_VOCAB = 32000.
// Core trick: logsumexp-matmul == max-shift + f16 WMMA GEMM with f32 accum.
//
// Round-2: `#pragma unroll 1` stopped the full unroll + t-loop hoist of the
// loop-invariant Pt fragments that was spilling to scratch.
// Round-3: widen to `#pragma unroll 4` so the scheduler can batch-issue the
// LDS + global fragment loads for 4 K-chunks ahead of 8 WMMAs, hiding the L2
// round-trip inside each block while staying at ~130 live VGPRs (no spills,
// and loads still live in a loop so they cannot be hoisted out of the t-loop).
// ---------------------------------------------------------------------------

#define N_STATES 512
#define M_VOCAB  32000
#define BATCH    64
#define T_STEPS  256

typedef __attribute__((ext_vector_type(16))) _Float16 v16h;
typedef __attribute__((ext_vector_type(8)))  _Float16 v8h;
typedef __attribute__((ext_vector_type(8)))  float    v8f;

// ---------------------------------------------------------------------------
// Prep kernel 1: log_softmax of priors.  pls[n] = priors[n] - lse(priors)
// One block of 512 threads.
// ---------------------------------------------------------------------------
__global__ void hmm_priors_kernel(const float* __restrict__ priors,
                                  float* __restrict__ pls) {
    __shared__ float red[N_STATES];
    const int tid = threadIdx.x;
    float v = priors[tid];

    red[tid] = v;
    __syncthreads();
    for (int s = N_STATES / 2; s > 0; s >>= 1) {
        if (tid < s) red[tid] = fmaxf(red[tid], red[tid + s]);
        __syncthreads();
    }
    const float mx = red[0];
    __syncthreads();

    red[tid] = __expf(v - mx);
    __syncthreads();
    for (int s = N_STATES / 2; s > 0; s >>= 1) {
        if (tid < s) red[tid] += red[tid + s];
        __syncthreads();
    }
    const float lse = mx + __logf(red[0]);
    pls[tid] = v - lse;
}

// ---------------------------------------------------------------------------
// Prep kernel 2: e0[b][n] = log_softmax(emit[:, x[b,0]], over states)[n].
// Only the 64 used vocab columns are ever touched (128 KB vs 64 MB).
// Grid = 64 blocks (one per batch row), 512 threads.
// ---------------------------------------------------------------------------
__global__ void hmm_e0_kernel(const float* __restrict__ emit,
                              const int* __restrict__ x,
                              float* __restrict__ e0) {
    __shared__ float red[N_STATES];
    const int b   = blockIdx.x;
    const int tid = threadIdx.x;
    const int m   = x[b * T_STEPS];           // x[b, 0]
    const float v = emit[(size_t)tid * M_VOCAB + m];

    red[tid] = v;
    __syncthreads();
    for (int s = N_STATES / 2; s > 0; s >>= 1) {
        if (tid < s) red[tid] = fmaxf(red[tid], red[tid + s]);
        __syncthreads();
    }
    const float mx = red[0];
    __syncthreads();

    red[tid] = __expf(v - mx);
    __syncthreads();
    for (int s = N_STATES / 2; s > 0; s >>= 1) {
        if (tid < s) red[tid] += red[tid + s];
        __syncthreads();
    }
    const float lse = mx + __logf(red[0]);
    e0[b * N_STATES + tid] = v - lse;
}

// ---------------------------------------------------------------------------
// Prep kernel 3: Pt[j][i] = exp(log_softmax(trans[:, j], over i))  in f16.
// Stored TRANSPOSED (row j = one transition-target column) so that a WMMA
// B-fragment is a single contiguous 32-byte load per lane.
// Grid = 512 blocks (one per column j), 512 threads.
// ---------------------------------------------------------------------------
__global__ void hmm_pt_kernel(const float* __restrict__ trans,
                              _Float16* __restrict__ Pt) {
    __shared__ float red[N_STATES];
    const int j   = blockIdx.x;
    const int tid = threadIdx.x;                       // = state i
    const float v = trans[(size_t)tid * N_STATES + j]; // trans[i, j]

    red[tid] = v;
    __syncthreads();
    for (int s = N_STATES / 2; s > 0; s >>= 1) {
        if (tid < s) red[tid] = fmaxf(red[tid], red[tid + s]);
        __syncthreads();
    }
    const float mx = red[0];
    __syncthreads();

    const float e = __expf(v - mx);
    red[tid] = e;
    __syncthreads();
    for (int s = N_STATES / 2; s > 0; s >>= 1) {
        if (tid < s) red[tid] += red[tid + s];
        __syncthreads();
    }
    const float inv = 1.0f / red[0];
    Pt[(size_t)j * N_STATES + tid] = (_Float16)(e * inv);  // in [0,1]: f16-safe
}

// ---------------------------------------------------------------------------
// Recurrence kernel. Grid = 4 WGs x 512 threads (16 wave32s).
// WG g owns batch rows b0 = 16*g .. 16*g+15  (M = 16 == WMMA tile height).
// Each step:
//   phase D: per-row (per-wave) max -> w = exp(alpha-m) (f16, A-frag layout)
//            -> rowsum -> ls = m + log(s); emit output at t == T[b]-1.
//   phase C: acc = w @ P via v_wmma_f32_16x16x32_f16 (16 K-chunks, 2 tiles
//            per wave); alpha <- e0 + m + log(acc).
// ---------------------------------------------------------------------------
__global__ void __launch_bounds__(512)
hmm_recurrence_kernel(const _Float16* __restrict__ Pt,
                      const float* __restrict__ e0,
                      const float* __restrict__ pls,
                      const int* __restrict__ Tlen,
                      float* __restrict__ out) {
    __shared__ __align__(32) float    alpha_s[16 * N_STATES];  // 32 KB
    __shared__ __align__(32) _Float16 w_s[16 * N_STATES];      // 16 KB
    __shared__ float mrow_s[16];

    const int tid  = threadIdx.x;
    const int wave = tid >> 5;
    const int lane = tid & 31;
    const int b0   = blockIdx.x * 16;

    // ---- init: alpha0 = e0 + priors ----
    for (int idx = tid; idx < 16 * N_STATES; idx += 512) {
        const int r = idx >> 9;          // /512
        const int n = idx & (N_STATES - 1);
        alpha_s[idx] = e0[(b0 + r) * N_STATES + n] + pls[n];
    }
    __syncthreads();

    const int nloc = lane & 15;          // N within tile / A-matrix row
    const int hi   = lane >> 4;          // lane-group select
    const int jt0  = wave * 2;           // this wave's two column tiles
    const int jt1  = wave * 2 + 1;
    const int myT0 = Tlen[b0 + wave] - 1;  // wave r owns row r's reductions

    // Per-lane invariant pieces of the fragment addresses.
    const _Float16* __restrict__ a_base = &w_s[nloc * N_STATES + (hi << 4)];
    const _Float16* __restrict__ b_base0 =
        &Pt[(size_t)(jt0 * 16 + nloc) * N_STATES + (hi << 4)];
    const _Float16* __restrict__ b_base1 =
        &Pt[(size_t)(jt1 * 16 + nloc) * N_STATES + (hi << 4)];

    for (int t = 0; t < T_STEPS; ++t) {
        // ---------- phase D: wave-local row reductions ----------
        const int r = wave;
        float mx = -INFINITY;
        for (int c = lane; c < N_STATES; c += 32)
            mx = fmaxf(mx, alpha_s[r * N_STATES + c]);
        for (int off = 16; off > 0; off >>= 1)
            mx = fmaxf(mx, __shfl_xor(mx, off, 32));

        float s = 0.0f;
        for (int c = lane; c < N_STATES; c += 32) {
            const float e = __expf(alpha_s[r * N_STATES + c] - mx);
            w_s[r * N_STATES + c] = (_Float16)e;   // in [0,1]: f16-safe
            s += e;
        }
        for (int off = 16; off > 0; off >>= 1)
            s += __shfl_xor(s, off, 32);

        if (lane == 0) {
            mrow_s[r] = mx;
            if (myT0 == t) out[b0 + r] = mx + __logf(s);  // log p(x) at len T[b]
        }

        if (t == T_STEPS - 1) break;
        __syncthreads();   // w, mrow visible to all waves

        // ---------- phase C: acc = w @ P  (f16 WMMA, f32 accum) ----------
        v8f acc0 = {};
        v8f acc1 = {};
        // unroll 4: scheduler batch-issues 4 chunks of fragment loads ahead
        // of 8 WMMAs (hides L2 latency); ~130 live VGPRs -> no spills, and
        // the loads remain inside a loop so they can't be hoisted/unrolled
        // out of the t-loop (the round-1 scratch-spill failure mode).
        #pragma unroll 4
        for (int kc = 0; kc < N_STATES; kc += 32) {
            // A fragment (16x32 f16) from LDS, contiguous k order (same as B):
            //   lanes 0-15 : M=lane,    K = kc+0..15
            //   lanes16-31 : M=lane-16, K = kc+16..31
            union { v16h v; v8h h[2]; } a;
            a.h[0] = *(const v8h*)(a_base + kc);
            a.h[1] = *(const v8h*)(a_base + kc + 8);

            // B fragments (32x16 f16): lanes 0-15 hold K=kc..kc+15,
            // lanes 16-31 hold K=kc+16..kc+31 -> one contiguous 32B load
            // from the transposed, L2-resident P.
            const v16h bfrag0 = *(const v16h*)(b_base0 + kc);
            const v16h bfrag1 = *(const v16h*)(b_base1 + kc);

            acc0 = __builtin_amdgcn_wmma_f32_16x16x32_f16(
                false, a.v, false, bfrag0, (short)0, acc0, false, false);
            acc1 = __builtin_amdgcn_wmma_f32_16x16x32_f16(
                false, a.v, false, bfrag1, (short)0, acc1, false, false);
        }

        // C layout: vgpr v -> row v + 8*hi, col = lane&15.
        // alpha_{t+1}[b,j] = e0[b,j] + m_b + log(acc[b,j])
        for (int v = 0; v < 8; ++v) {
            const int row = v + (hi ? 8 : 0);
            const float mrow = mrow_s[row];
            const int j0 = jt0 * 16 + nloc;
            const int j1 = jt1 * 16 + nloc;
            alpha_s[row * N_STATES + j0] =
                e0[(b0 + row) * N_STATES + j0] + mrow + __logf(acc0[v]);
            alpha_s[row * N_STATES + j1] =
                e0[(b0 + row) * N_STATES + j1] + mrow + __logf(acc1[v]);
        }
        __syncthreads();   // alpha ready for next step's phase D
    }
}

// ---------------------------------------------------------------------------
// Host launcher
// ---------------------------------------------------------------------------
extern "C" void kernel_launch(void* const* d_in, const int* in_sizes, int n_in,
                              void* d_out, int out_size, void* d_ws, size_t ws_size,
                              hipStream_t stream) {
    const int*   x      = (const int*)d_in[0];     // (64, 256)
    const int*   Tlen   = (const int*)d_in[1];     // (64,)
    const float* priors = (const float*)d_in[2];   // (512,)
    const float* trans  = (const float*)d_in[3];   // (512, 512)
    const float* emit   = (const float*)d_in[4];   // (512, 32000)
    float*       out    = (float*)d_out;           // (64, 1) f32

    // Workspace layout (hipMalloc base => 256B aligned):
    //   Pt  : 512*512 f16  = 512 KB   (transposed transition probabilities)
    //   e0  : 64*512  f32  = 128 KB
    //   pls : 512     f32  =   2 KB
    char* ws = (char*)d_ws;
    _Float16* Pt  = (_Float16*)ws;
    float*    e0  = (float*)(ws + (size_t)N_STATES * N_STATES * sizeof(_Float16));
    float*    pls = (float*)(ws + (size_t)N_STATES * N_STATES * sizeof(_Float16)
                                + (size_t)BATCH * N_STATES * sizeof(float));

    hmm_priors_kernel<<<1, N_STATES, 0, stream>>>(priors, pls);
    hmm_e0_kernel<<<BATCH, N_STATES, 0, stream>>>(emit, x, e0);
    hmm_pt_kernel<<<N_STATES, N_STATES, 0, stream>>>(trans, Pt);
    hmm_recurrence_kernel<<<BATCH / 16, 512, 0, stream>>>(Pt, e0, pls, Tlen, out);
}
